// MambaDecoderBlock_64768106824322
// MI455X (gfx1250) — compile-verified, matrix operand order checked by reading, not run on hardware
//
#include <hip/hip_runtime.h>
#include <math.h>

typedef __attribute__((ext_vector_type(16))) __bf16 v16bf;
typedef __attribute__((ext_vector_type(8)))  float  v8f;
typedef unsigned int   u32;
typedef unsigned short u16;

#define BDIM  256
#define LSEQ  4096
#define DIN   512
#define DST   16

union FragU { u32 u[8]; v16bf v; };

__device__ __forceinline__ u16 f2bf(float v) {
    union { __bf16 h; u16 s; } c;
    c.h = (__bf16)v;          // v_cvt bf16, RNE
    return c.s;
}

// async global -> LDS 16-byte copy (CDNA5, tracked by ASYNCcnt, in-order done)
__device__ __forceinline__ void async_cp16(u32 lds_addr, const void* gptr) {
    asm volatile("global_load_async_to_lds_b128 %0, %1, off"
                 :: "v"(lds_addr), "v"(gptr) : "memory");
}
__device__ __forceinline__ void async_wait(int n_uniform) {
    // n_uniform is block-uniform (0, 2 or 4): allow that many newer copies in flight
    if (n_uniform == 4)      asm volatile("s_wait_asynccnt 0x4" ::: "memory");
    else if (n_uniform == 2) asm volatile("s_wait_asynccnt 0x2" ::: "memory");
    else                     asm volatile("s_wait_asynccnt 0x0" ::: "memory");
}

// ---------------------------------------------------------------------------
// 0) fp32 -> bf16 buffer conversion (weights, once per call)
// ---------------------------------------------------------------------------
__global__ void k_f32_bf16(const float* __restrict__ src, u16* __restrict__ dst, int n) {
    int i = blockIdx.x * blockDim.x + threadIdx.x;
    if (i < n) dst[i] = f2bf(src[i]);
}

// ---------------------------------------------------------------------------
// 1) 2x2 transposed-conv upsample (256->128) + concat skip -> xc (NCHW, f32)
//    and seq_bf (B*L, 256) bf16 for in_proj A operand.
// ---------------------------------------------------------------------------
__global__ void k_upsample_concat(const float* __restrict__ x,
                                  const float* __restrict__ skip,
                                  const float* __restrict__ up_w,
                                  const float* __restrict__ up_b,
                                  float* __restrict__ xc,
                                  u16* __restrict__ seqbf) {
    int idx = blockIdx.x * blockDim.x + threadIdx.x;     // B*256*4096
    int hw = idx & 4095;
    int h2 = hw >> 6, w2 = hw & 63;
    int c2 = (idx >> 12) & 255;
    int b  = idx >> 20;
    float v;
    if (c2 < 128) {
        int hh = h2 >> 1, ww = w2 >> 1;
        int sub = (h2 & 1) * 2 + (w2 & 1);
        const float* xp = x + ((size_t)b << 18) + hh * 32 + ww;   // b*256*1024
        const float* wp = up_w + c2 * 4 + sub;                     // [c][c2][sub]
        float s = up_b[c2];
        for (int c = 0; c < 256; ++c)
            s += xp[c << 10] * wp[c << 9];
        v = s;
    } else {
        v = skip[((size_t)b * 128 + (c2 - 128)) * 4096 + hw];
    }
    xc[idx] = v;
    seqbf[((size_t)b * 4096 + hw) * 256 + c2] = f2bf(v);
}

// ---------------------------------------------------------------------------
// 2) WMMA bf16 GEMM:  C[M,N] = act( A[M,K] * W[N,K]^T + bias[N] )
//    A, W bf16 (u16 bits).  Block: 256 thr = 8 waves; tile 128x128; wave ->
//    32x64 (2x4 wmma).  Double-buffered LDS: async global->LDS b128 copies
//    for tile kt+1 overlap the 8 WMMAs of tile kt; in-order ASYNCcnt wait
//    with limit = copies in flight for the next tile.
//    Slow path (edge tiles): clamped loads + select.  act: 0 none, 1 softplus.
// ---------------------------------------------------------------------------
__global__ void k_gemm_bf16(const u16* __restrict__ A, int lda,
                            const u16* __restrict__ W,
                            const float* __restrict__ bias,
                            float* __restrict__ C,
                            u16* __restrict__ Cbf,
                            int M, int N, int K, int act) {
    __shared__ u32 As[2][128][20];   // per buffer: 16 data u32 (32 bf16) + 4 pad
    __shared__ u32 Ws[2][128][20];

    const int tid  = threadIdx.x;
    const int bm   = blockIdx.x * 128;
    const int bn   = blockIdx.y * 128;
    const int wave = tid >> 5;
    const int lane = tid & 31;
    const int wm   = (wave >> 1) * 32;      // wave tile row in block
    const int wn   = (wave & 1) * 64;       // wave tile col in block
    const int lrow = lane & 15;
    const int kb4  = (lane >> 4) * 4;       // u32 base within K-half (ISA layout)

    const int r    = tid >> 1;              // staging row 0..127
    const int cb16 = (tid & 1) * 16;        // staging bf16-col base (0 or 16)
    const int cu   = cb16 >> 1;             // staging u32-col base (0 or 8)

    const bool aFull = (bm + 128 <= M);
    const bool bFull = (bn + 128 <= N);
    const int kTiles = (K + 31) >> 5;

    // stage tile kt into LDS buffer `buf`; returns #async copies issued (uniform)
    auto stage = [&](int kt, int buf) -> int {
        const int k0 = kt << 5;
        const bool kfull = (k0 + 32 <= K);
        int cnt = 0;
        // ---- A tile (128 rows x 32 bf16) ----
        if (kfull && aFull) {
            const u16* g = A + (size_t)(bm + r) * lda + (k0 + cb16);
            u32 lds = (u32)(size_t)&As[buf][r][cu];
            async_cp16(lds,      g);
            async_cp16(lds + 16, g + 8);
            cnt += 2;
        } else {
            const int gm  = bm + r;
            const int gmc = gm < M ? gm : M - 1;
            u32* dst = &As[buf][r][cu];
            #pragma unroll
            for (int i = 0; i < 16; i += 2) {
                int ka = k0 + cb16 + i, kb = ka + 1;
                u32 va = A[(size_t)gmc * lda + (ka < K ? ka : K - 1)];
                u32 vb = A[(size_t)gmc * lda + (kb < K ? kb : K - 1)];
                if (gm >= M || ka >= K) va = 0;
                if (gm >= M || kb >= K) vb = 0;
                dst[i >> 1] = va | (vb << 16);
            }
        }
        // ---- W tile (128 rows x 32 bf16) ----
        if (kfull && bFull) {
            const u16* g = W + (size_t)(bn + r) * K + (k0 + cb16);
            u32 lds = (u32)(size_t)&Ws[buf][r][cu];
            async_cp16(lds,      g);
            async_cp16(lds + 16, g + 8);
            cnt += 2;
        } else {
            const int gn  = bn + r;
            const int gnc = gn < N ? gn : N - 1;
            u32* dst = &Ws[buf][r][cu];
            #pragma unroll
            for (int i = 0; i < 16; i += 2) {
                int ka = k0 + cb16 + i, kb = ka + 1;
                u32 va = W[(size_t)gnc * K + (ka < K ? ka : K - 1)];
                u32 vb = W[(size_t)gnc * K + (kb < K ? kb : K - 1)];
                if (gn >= N || ka >= K) va = 0;
                if (gn >= N || kb >= K) vb = 0;
                dst[i >> 1] = va | (vb << 16);
            }
        }
        return cnt;
    };

    const v8f zero = {};
    v8f acc[2][4];
    #pragma unroll
    for (int i = 0; i < 2; ++i)
        #pragma unroll
        for (int j = 0; j < 4; ++j) acc[i][j] = zero;

    stage(0, 0);                               // prologue: tile 0 -> buffer 0

    for (int kt = 0; kt < kTiles; ++kt) {
        int cnt_next = 0;
        if (kt + 1 < kTiles)
            cnt_next = stage(kt + 1, (kt + 1) & 1);   // prefetch next tile
        async_wait(cnt_next);                  // tile kt landed; next stays in flight
        __syncthreads();

        const int buf = kt & 1;
        // ---- fragments per ISA 16-bit layout: lanes 0-15 K 0..7/16..23,
        //      lanes 16-31 K 8..15/24..31 ----
        FragU fa[2], fb[4];
        #pragma unroll
        for (int mt = 0; mt < 2; ++mt) {
            const u32* ap = &As[buf][wm + mt * 16 + lrow][0];
            #pragma unroll
            for (int i = 0; i < 4; ++i) {
                fa[mt].u[i]     = ap[kb4 + i];
                fa[mt].u[4 + i] = ap[8 + kb4 + i];
            }
        }
        #pragma unroll
        for (int nt = 0; nt < 4; ++nt) {
            const u32* bp = &Ws[buf][wn + nt * 16 + lrow][0];
            #pragma unroll
            for (int i = 0; i < 4; ++i) {
                fb[nt].u[i]     = bp[kb4 + i];
                fb[nt].u[4 + i] = bp[8 + kb4 + i];
            }
        }
        #pragma unroll
        for (int mt = 0; mt < 2; ++mt)
            #pragma unroll
            for (int nt = 0; nt < 4; ++nt)
                acc[mt][nt] = __builtin_amdgcn_wmma_f32_16x16x32_bf16(
                    false, fa[mt].v, false, fb[nt].v,
                    (short)0, acc[mt][nt], false, false);
        __syncthreads();   // don't let iter kt+1 overwrite a buffer still being read
    }

    // ---- epilogue: C/D layout lane 0-15: M=r, lane 16-31: M=8+r; N=lane&15
    #pragma unroll
    for (int mt = 0; mt < 2; ++mt) {
        const int mb = bm + wm + mt * 16 + ((lane >> 4) << 3);
        #pragma unroll
        for (int nt = 0; nt < 4; ++nt) {
            const int n = bn + wn + nt * 16 + lrow;
            const float* accp = (const float*)&acc[mt][nt];
            float bv = (bias && n < N) ? bias[n] : 0.0f;
            #pragma unroll
            for (int rr = 0; rr < 8; ++rr) {
                int m = mb + rr;
                if (m < M && n < N) {
                    float v = accp[rr] + bv;
                    if (act == 1) v = (v > 20.0f) ? v : log1pf(expf(v));  // softplus
                    C[(size_t)m * N + n] = v;
                    if (Cbf) Cbf[(size_t)m * N + n] = f2bf(v);
                }
            }
        }
    }
}

// ---------------------------------------------------------------------------
// 3) causal depthwise conv1d (k=4) over L + SiLU.  xi = xz[:, :512] (ld=1024)
//    writes u (f32, for scan) and u_bf (bf16, x_proj A operand)
// ---------------------------------------------------------------------------
__global__ void k_conv1d_silu(const float* __restrict__ xz,
                              const float* __restrict__ w,
                              const float* __restrict__ bias,
                              float* __restrict__ u,
                              u16* __restrict__ ubf) {
    int idx = blockIdx.x * blockDim.x + threadIdx.x;   // 8192*512
    int d  = idx & 511;
    int bl = idx >> 9;
    int l  = bl & 4095, b = bl >> 12;
    float s = bias[d];
    #pragma unroll
    for (int j = 0; j < 4; ++j) {
        int ll = l - 3 + j;
        if (ll >= 0)
            s += xz[((size_t)(b * 4096 + ll) << 10) + d] * w[d * 4 + j];
    }
    s = s / (1.0f + __expf(-s));
    u[idx]   = s;
    ubf[idx] = f2bf(s);
}

// ---------------------------------------------------------------------------
// 4) selective scan: channel-parallel, 16-state h in registers per thread.
//    Fused: +u*D, SiLU(z) gating.  Writes yg_bf (bf16, out_proj A operand).
// ---------------------------------------------------------------------------
__global__ void k_scan(const float* __restrict__ dt,
                       const float* __restrict__ dbc,    // (B*L,48): [16:32)=B, [32:48)=C
                       const float* __restrict__ u,
                       const float* __restrict__ xz,     // z at col 512+d, ld=1024
                       const float* __restrict__ A_log,
                       const float* __restrict__ Dp,
                       u16* __restrict__ ygbf) {
    const int b = blockIdx.x;     // batch
    const int d = threadIdx.x;    // channel 0..511
    float Ar[DST], h[DST];
    #pragma unroll
    for (int n = 0; n < DST; ++n) {
        Ar[n] = -__expf(A_log[d * DST + n]);
        h[n]  = 0.0f;
    }
    const float Dv = Dp[d];
    for (int l = 0; l < LSEQ; ++l) {
        const size_t bl = (size_t)b * LSEQ + l;
        if (l + 1 < LSEQ) {
            __builtin_prefetch(dt + (bl + 1) * DIN + d, 0, 0);
            __builtin_prefetch(dbc + (bl + 1) * 48 + 16, 0, 0);
        }
        const float dtv = dt[bl * DIN + d];
        const float uv  = u[bl * DIN + d];
        const float du  = dtv * uv;
        // B (16) then C (16): 32 contiguous, 16B-aligned floats
        union { float4 v4[8]; float f[32]; } bc;
        const float4* bptr = (const float4*)(dbc + bl * 48 + 16);
        #pragma unroll
        for (int i = 0; i < 8; ++i) bc.v4[i] = bptr[i];
        float y = 0.0f;
        #pragma unroll
        for (int n = 0; n < DST; ++n) {
            const float dA = __expf(dtv * Ar[n]);
            h[n] = dA * h[n] + du * bc.f[n];
            y += h[n] * bc.f[16 + n];
        }
        y += uv * Dv;
        const float zv = xz[bl * 1024 + DIN + d];
        y *= zv / (1.0f + __expf(-zv));
        ygbf[bl * DIN + d] = f2bf(y);
    }
}

// ---------------------------------------------------------------------------
// 5) residual: xr(NCHW,f32) = xc + mam(B*L,256) transposed
// ---------------------------------------------------------------------------
__global__ void k_residual(const float* __restrict__ xc,
                           const float* __restrict__ mam,
                           float* __restrict__ xr) {
    int idx = blockIdx.x * blockDim.x + threadIdx.x;   // B*256*4096
    int hw = idx & 4095;
    int c  = (idx >> 12) & 255;
    int b  = idx >> 20;
    xr[idx] = xc[idx] + mam[((size_t)(b * 4096 + hw)) * 256 + c];
}

// ---------------------------------------------------------------------------
// 6) im2col (bf16) for 3x3 pad-1 conv: col(B*L, 2304), j = c*9 + ky*3 + kx
// ---------------------------------------------------------------------------
__global__ void k_im2col(const float* __restrict__ xr, u16* __restrict__ col) {
    const int m = blockIdx.x;            // b*4096 + hw
    const int b = m >> 12, hw = m & 4095;
    const int h = hw >> 6, w = hw & 63;
    u16* dst = col + (size_t)m * 2304;
    const float* src = xr + ((size_t)b << 20);   // b*256*4096
    #pragma unroll
    for (int i = 0; i < 9; ++i) {
        int j = threadIdx.x + (i << 8);
        int c = j / 9, r9 = j % 9;
        int ky = r9 / 3, kx = r9 % 3;
        int hh = h + ky - 1, ww = w + kx - 1;
        float v = 0.0f;
        if (hh >= 0 && hh < 64 && ww >= 0 && ww < 64)
            v = src[((size_t)c << 12) + hh * 64 + ww];
        dst[j] = f2bf(v);
    }
}

// ---------------------------------------------------------------------------
// 7) BN stats over GEMM-layout y2t (B*L, 128): per-channel mean/var
// ---------------------------------------------------------------------------
__global__ void k_bnstats(const float* __restrict__ y2t, float* __restrict__ stats) {
    const int o = blockIdx.x, tid = threadIdx.x;
    float s = 0.0f, ss = 0.0f;
    for (int m = tid; m < 8192; m += 256) {
        float v = y2t[(size_t)m * 128 + o];
        s += v; ss += v * v;
    }
    __shared__ float sh[256], sh2[256];
    sh[tid] = s; sh2[tid] = ss;
    __syncthreads();
    for (int st = 128; st > 0; st >>= 1) {
        if (tid < st) { sh[tid] += sh[tid + st]; sh2[tid] += sh2[tid + st]; }
        __syncthreads();
    }
    if (tid == 0) {
        float mu = sh[0] * (1.0f / 8192.0f);
        float var = sh2[0] * (1.0f / 8192.0f) - mu * mu;
        stats[o * 2] = mu;
        stats[o * 2 + 1] = var;
    }
}

// ---------------------------------------------------------------------------
// 8) BN apply + exact GELU; transpose (B*L,128) -> NCHW out
// ---------------------------------------------------------------------------
__global__ void k_bn_gelu(const float* __restrict__ y2t,
                          const float* __restrict__ stats,
                          const float* __restrict__ gamma,
                          const float* __restrict__ beta,
                          float* __restrict__ out) {
    int idx = blockIdx.x * blockDim.x + threadIdx.x;   // NCHW: 2*128*4096
    int hw = idx & 4095;
    int o  = (idx >> 12) & 127;
    int b  = idx >> 19;
    float mu = stats[o * 2], var = stats[o * 2 + 1];
    float raw = y2t[(size_t)(b * 4096 + hw) * 128 + o];
    float v = (raw - mu) * rsqrtf(var + 1e-5f) * gamma[o] + beta[o];
    out[idx] = 0.5f * v * (1.0f + erff(v * 0.70710678118654752f));
}

// ---------------------------------------------------------------------------
extern "C" void kernel_launch(void* const* d_in, const int* in_sizes, int n_in,
                              void* d_out, int out_size, void* d_ws, size_t ws_size,
                              hipStream_t stream) {
    const float* x         = (const float*)d_in[0];
    const float* skip      = (const float*)d_in[1];
    const float* up_w      = (const float*)d_in[2];
    const float* up_b      = (const float*)d_in[3];
    const float* in_proj_w = (const float*)d_in[4];
    const float* conv1d_w  = (const float*)d_in[5];
    const float* conv1d_b  = (const float*)d_in[6];
    const float* x_proj_w  = (const float*)d_in[7];
    const float* dt_proj_w = (const float*)d_in[8];
    const float* dt_proj_b = (const float*)d_in[9];
    const float* A_log     = (const float*)d_in[10];
    const float* Dcoef     = (const float*)d_in[11];
    const float* out_proj_w= (const float*)d_in[12];
    const float* conv_w    = (const float*)d_in[13];
    const float* conv_b    = (const float*)d_in[14];
    const float* bn_gamma  = (const float*)d_in[15];
    const float* bn_beta   = (const float*)d_in[16];
    float* out = (float*)d_out;

    float* ws = (float*)d_ws;
    // workspace layout (units = floats); bf16 buffers use half a float/elem
    float* p_xc    = ws;                   // 2,097,152 f  xc (NCHW f32)      [.. residual]
    u16*   p_seqbf = (u16*)(ws + 2097152); // 2,097,152 bf (=1,048,576 f)     [.. in_proj]
    float* p_xz    = ws + 3145728;         // 8,388,608 f  xz (B*L,1024)      [.. scan]
    float* p_u     = ws + 11534336;        // 4,194,304 f  u f32              [.. scan]
    u16*   p_ubf   = (u16*)(ws + 15728640);// 4,194,304 bf (=2,097,152 f)     [.. x_proj]
    float* p_dbc   = ws + 17825792;        //   393,216 f  dbc f32            [.. scan]
    u16*   p_dbcbf = (u16*)(ws + 18219008);//   393,216 bf (=196,608 f)       [.. dt_proj]
    float* p_dt    = ws + 18415616;        // 4,194,304 f  dt f32             [.. scan]
    u16*   p_ygbf  = (u16*)(ws + 22609920);// 4,194,304 bf (=2,097,152 f)     [.. out_proj]
    u16*   p_wbf   = (u16*)(ws + 24707072);//   720,896 bf weights (whole call)
    // reuse regions:
    float* p_mam   = p_u;                  // (u dead after scan)             [.. residual]
    float* p_xr    = p_xz;                 // (xz dead after scan)            [.. im2col]
    u16*   p_colbf = (u16*)(ws + 5242880); // 18,874,368 bf (=9,437,184 f), dead regions
    float* p_y2    = ws;                   // 1,048,576 f (xc dead after residual)
    float* p_stat  = ws + 1048576;         //       256 f

    // bf16 weight sub-buffers
    u16* w_inproj  = p_wbf;                // 262,144
    u16* w_xproj   = p_wbf + 262144;       //  24,576
    u16* w_dtproj  = p_wbf + 286720;       //   8,192
    u16* w_outproj = p_wbf + 294912;       // 131,072
    u16* w_conv    = p_wbf + 425984;       // 294,912

    // 0) convert weights to bf16
    k_f32_bf16<<<1024, BDIM, 0, stream>>>(in_proj_w,  w_inproj,  262144);
    k_f32_bf16<<<  96, BDIM, 0, stream>>>(x_proj_w,   w_xproj,    24576);
    k_f32_bf16<<<  32, BDIM, 0, stream>>>(dt_proj_w,  w_dtproj,    8192);
    k_f32_bf16<<< 512, BDIM, 0, stream>>>(out_proj_w, w_outproj, 131072);
    k_f32_bf16<<<1152, BDIM, 0, stream>>>(conv_w,     w_conv,    294912);

    // 1) upsample + concat
    k_upsample_concat<<<2097152 / BDIM, BDIM, 0, stream>>>(x, skip, up_w, up_b, p_xc, p_seqbf);

    // 2) in_proj:  (8192,1024,K=256)
    k_gemm_bf16<<<dim3(64, 8), BDIM, 0, stream>>>(
        p_seqbf, 256, w_inproj, nullptr, p_xz, nullptr, 8192, 1024, 256, 0);

    // 3) depthwise conv1d + SiLU
    k_conv1d_silu<<<(8192 * 512) / BDIM, BDIM, 0, stream>>>(p_xz, conv1d_w, conv1d_b, p_u, p_ubf);

    // 4) x_proj: (8192,48,K=512); also emit bf16 copy for dt_proj A
    k_gemm_bf16<<<dim3(64, 1), BDIM, 0, stream>>>(
        p_ubf, 512, w_xproj, nullptr, p_dbc, p_dbcbf, 8192, 48, 512, 0);

    // 5) dt_proj + softplus: (8192,512,K=16), A = dbc_bf[:, :16] (lda=48)
    k_gemm_bf16<<<dim3(64, 4), BDIM, 0, stream>>>(
        p_dbcbf, 48, w_dtproj, dt_proj_b, p_dt, nullptr, 8192, 512, 16, 1);

    // 6) selective scan + D skip + SiLU(z) gate -> bf16
    k_scan<<<2, 512, 0, stream>>>(p_dt, p_dbc, p_u, p_xz, A_log, Dcoef, p_ygbf);

    // 7) out_proj: (8192,256,K=512)
    k_gemm_bf16<<<dim3(64, 2), BDIM, 0, stream>>>(
        p_ygbf, 512, w_outproj, nullptr, p_mam, nullptr, 8192, 256, 512, 0);

    // 8) residual
    k_residual<<<2097152 / BDIM, BDIM, 0, stream>>>(p_xc, p_mam, p_xr);

    // 9) 3x3 conv as im2col(bf16) + WMMA GEMM (8192,128,K=2304)
    k_im2col<<<8192, BDIM, 0, stream>>>(p_xr, p_colbf);
    k_gemm_bf16<<<dim3(64, 1), BDIM, 0, stream>>>(
        p_colbf, 2304, w_conv, conv_b, p_y2, nullptr, 8192, 128, 2304, 0);

    // 10) batch-norm stats + apply + GELU
    k_bnstats<<<128, BDIM, 0, stream>>>(p_y2, p_stat);
    k_bn_gelu<<<1048576 / BDIM, BDIM, 0, stream>>>(p_y2, p_stat, bn_gamma, bn_beta, out);
}
